// Qwen3MoeSparseMoeBlock_89953795048057
// MI455X (gfx1250) — matrix operand order repaired and checked by reading, not compiled, not run on hardware
//
#include <hip/hip_runtime.h>

// ---------------- Problem constants (fixed by the reference) ----------------
#define K_TOP 8
constexpr int H_DIM = 2048;   // hidden
constexpr int I_DIM = 768;    // intermediate
constexpr int E_NUM = 64;     // experts

// GEMM tiling: 128 tokens x 64 cols per block, K-chunk 32 (one bf16 WMMA step)
constexpr int BM = 128;
constexpr int BN = 64;
constexpr int BK = 32;
// LDS tiles hold packed bf16 pairs (uint32). Row stride 20 uints = 80 bytes:
// multiple of 16B so every 4-uint run is ds_load_b128-able.
constexpr int LDA = BK / 2 + 4;   // 20

typedef __bf16 v16bf __attribute__((ext_vector_type(16)));
typedef __bf16 v2bf  __attribute__((ext_vector_type(2)));
typedef float  v2f   __attribute__((ext_vector_type(2)));
typedef float  v8f   __attribute__((ext_vector_type(8)));

union FragU { v16bf v; uint4 q[2]; };

// Two floats -> one packed bf16 pair via vector conversion: selects the packed
// v_cvt_pk_bf16_f32 (no half-merge movs); bit_cast is free.
__device__ __forceinline__ unsigned int pack_bf16x2(float lo, float hi) {
  v2f f; f[0] = lo; f[1] = hi;
  v2bf p = __builtin_convertvector(f, v2bf);
  return __builtin_bit_cast(unsigned int, p);
}

// ---------------- 1) Gating: logits -> softmax -> top-8 -> renorm ----------
__global__ __launch_bounds__(E_NUM) void gate_topk_kernel(
    const float* __restrict__ x, const float* __restrict__ gw,
    int* __restrict__ topk_idx, float* __restrict__ topk_w) {
  const int t = blockIdx.x;
  const int e = threadIdx.x;              // one thread per expert
  __shared__ float sl[E_NUM];
  const float* xr = x + (size_t)t * H_DIM;
  const float* gr = gw + (size_t)e * H_DIM;
  float acc = 0.f;
  for (int h = 0; h < H_DIM; h += 4) {
    float4 a = *(const float4*)(xr + h);
    float4 b = *(const float4*)(gr + h);
    acc += a.x * b.x + a.y * b.y + a.z * b.z + a.w * b.w;
  }
  sl[e] = acc;
  __syncthreads();
  if (e == 0) {
    float mx = sl[0];
#pragma clang loop unroll(disable)
    for (int i = 1; i < E_NUM; ++i) mx = fmaxf(mx, sl[i]);
    float sum = 0.f;
#pragma clang loop unroll(disable)
    for (int i = 0; i < E_NUM; ++i) { sl[i] = __expf(sl[i] - mx); sum += sl[i]; }
    const float inv = 1.f / sum;
    float wsum = 0.f;
    int   idx[K_TOP]; float wv[K_TOP];
#pragma clang loop unroll(disable)
    for (int k = 0; k < K_TOP; ++k) {
      int best = 0; float bv = -1.f;
#pragma clang loop unroll(disable)
      for (int i = 0; i < E_NUM; ++i) {
        float p = sl[i] * inv;
        if (p > bv) { bv = p; best = i; }
      }
      idx[k] = best; wv[k] = bv; wsum += bv;
      sl[best] = -1.f;                    // exclude from further rounds
    }
    const float rn = 1.f / (wsum + 1e-20f);
#pragma clang loop unroll(disable)
    for (int k = 0; k < K_TOP; ++k) {
      topk_idx[t * K_TOP + k] = idx[k];
      topk_w[t * K_TOP + k]   = wv[k] * rn;
    }
  }
}

// ---------------- 2) Bucketing: counts -> offsets -> scatter ----------------
__global__ void count_kernel(const int* __restrict__ topk_idx,
                             int* __restrict__ counts, int n) {
  int i = blockIdx.x * blockDim.x + threadIdx.x;
  if (i < n) atomicAdd(&counts[topk_idx[i]], 1);
}

__global__ void offsets_kernel(const int* __restrict__ counts,
                               int* __restrict__ offsets,
                               int* __restrict__ cursors) {
  if (blockIdx.x == 0 && threadIdx.x == 0) {
    int acc = 0;
#pragma clang loop unroll(disable)
    for (int e = 0; e < E_NUM; ++e) {
      offsets[e] = acc; cursors[e] = acc; acc += counts[e];
    }
    offsets[E_NUM] = acc;
  }
}

__global__ void scatter_kernel(const int* __restrict__ topk_idx,
                               const float* __restrict__ topk_w,
                               int* __restrict__ cursors,
                               int* __restrict__ bucket_tok,
                               float* __restrict__ bucket_w, int n) {
  int i = blockIdx.x * blockDim.x + threadIdx.x;
  if (i < n) {
    int eid = topk_idx[i];
    int pos = atomicAdd(&cursors[eid], 1);
    bucket_tok[pos] = i / K_TOP;
    bucket_w[pos]   = topk_w[i];
  }
}

// -------- Fragment builders (packed-pair LDS, all ds_load_b128-friendly) ----
// A fragment: VGPR r holds K-pair ((r<4)?r:8+(r-4)) + 4*(lane>=16)  -> two
// contiguous 4-uint runs per lane.
__device__ __forceinline__ v16bf load_a_frag(const unsigned int* aRow, int lane) {
  const int ab = (lane >> 4) << 2;        // 0 or 4 (K-pair half offset)
  FragU u;
  u.q[0] = *(const uint4*)(aRow + ab);
  u.q[1] = *(const uint4*)(aRow + ab + 8);
  return u.v;
}
// B fragment (from transposed tile sBt[n][kpair]): same K-pair order as A
// -> two contiguous 4-uint runs per lane.
__device__ __forceinline__ v16bf load_b_frag(const unsigned int* bRow, int lane) {
  const int kb2 = (lane >> 4) << 2;       // 0 or 4
  FragU u;
  u.q[0] = *(const uint4*)(bRow + kb2);
  u.q[1] = *(const uint4*)(bRow + kb2 + 8);
  return u.v;
}

// ---------------- 3) Grouped up/gate GEMM + SwiGLU (bf16 WMMA) -------------
// Block: 256 threads = 8 wave32. Wave w owns token rows [16w,16w+16) and all
// 4 n-subtiles (64 cols of I). K loop over H in 32-chunks.
__global__ __launch_bounds__(256) void moe_up_kernel(
    const float* __restrict__ x,  const float* __restrict__ w1,
    const float* __restrict__ w3, const int* __restrict__ offsets,
    const int* __restrict__ counts, const int* __restrict__ bucket_tok,
    unsigned short* __restrict__ act) {
  const int e    = blockIdx.z;
  const int n_e  = counts[e];
  const int row0 = blockIdx.x * BM;
  if (row0 >= n_e) return;                 // uniform across block
  const int cb  = blockIdx.y * BN;
  const int off = offsets[e];

  __shared__ unsigned int sA  [BM][LDA];   // 128 x 20 uints (bf16 pairs)
  __shared__ unsigned int sBt1[BN][LDA];   // transposed: [n][kpair]
  __shared__ unsigned int sBt3[BN][LDA];

  const int tid  = threadIdx.x;
  const int lane = tid & 31;
  const int wave = tid >> 5;

  // ---- Hoisted, loop-invariant source pointers -------------------------
  // A: thread owns rows (tid>>3)+32*it at float column c4 (gathered tokens).
  const int aRow0 = tid >> 3;
  const int c4    = (tid & 7) << 2;
  const float* aPtr[4];
#pragma unroll
  for (int it = 0; it < 4; ++it) {
    int grow = row0 + aRow0 + it * 32;
    if (grow < n_e) {
      int tok   = bucket_tok[off + grow];
      aPtr[it]  = x + (size_t)tok * H_DIM + c4;
    } else {
      aPtr[it]  = nullptr;
    }
  }
  // B: thread owns K-pair k2 and 4 n-cols starting at n4.
  const int k2 = tid >> 4;                 // 0..15
  const int n4 = (tid & 15) << 2;          // 0,4,..,60
  const float* b1p = w1 + ((size_t)e * H_DIM + 2 * k2) * I_DIM + cb + n4;
  const float* b3p = w3 + ((size_t)e * H_DIM + 2 * k2) * I_DIM + cb + n4;

  const v8f vzero = {0.f, 0.f, 0.f, 0.f, 0.f, 0.f, 0.f, 0.f};
  v8f accG[4], accU[4];
#pragma unroll
  for (int nt = 0; nt < 4; ++nt) { accG[nt] = vzero; accU[nt] = vzero; }

  for (int k0 = 0; k0 < H_DIM; k0 += BK) {
    // A tile: 128 x 32 fp32 -> packed bf16 pairs in LDS
#pragma unroll
    for (int it = 0; it < 4; ++it) {
      uint2 p;
      if (aPtr[it]) {
        float4 v = *(const float4*)(aPtr[it] + k0);
        p.x = pack_bf16x2(v.x, v.y);
        p.y = pack_bf16x2(v.z, v.w);
      } else {
        p.x = 0u; p.y = 0u;               // integer zero fill (no cvt chain)
      }
      *(uint2*)&sA[aRow0 + it * 32][c4 >> 1] = p;
    }
    // B tiles: w1/w3 32x64 fp32 -> TRANSPOSED [n][kpair] packed bf16
    {
      const float* p1 = b1p + (size_t)k0 * I_DIM;
      const float* p3 = b3p + (size_t)k0 * I_DIM;
      float4 r0 = *(const float4*)(p1);
      float4 r1 = *(const float4*)(p1 + I_DIM);
      sBt1[n4 + 0][k2] = pack_bf16x2(r0.x, r1.x);
      sBt1[n4 + 1][k2] = pack_bf16x2(r0.y, r1.y);
      sBt1[n4 + 2][k2] = pack_bf16x2(r0.z, r1.z);
      sBt1[n4 + 3][k2] = pack_bf16x2(r0.w, r1.w);
      float4 s0 = *(const float4*)(p3);
      float4 s1 = *(const float4*)(p3 + I_DIM);
      sBt3[n4 + 0][k2] = pack_bf16x2(s0.x, s1.x);
      sBt3[n4 + 1][k2] = pack_bf16x2(s0.y, s1.y);
      sBt3[n4 + 2][k2] = pack_bf16x2(s0.z, s1.z);
      sBt3[n4 + 3][k2] = pack_bf16x2(s0.w, s1.w);
      if (k0 + BK < H_DIM) {   // prefetch next K-chunk (global_prefetch_b8)
        __builtin_prefetch(p1 + (size_t)BK * I_DIM, 0, 0);
        __builtin_prefetch(p3 + (size_t)BK * I_DIM, 0, 0);
      }
    }
    __syncthreads();

    // Load all fragments, then issue the 8 WMMAs back-to-back.
    v16bf a = load_a_frag(sA[(wave << 4) + (lane & 15)], lane);
    const int nb = lane & 15;
    v16bf bf1[4], bf3[4];
#pragma unroll
    for (int nt = 0; nt < 4; ++nt) {
      bf1[nt] = load_b_frag(sBt1[(nt << 4) + nb], lane);
      bf3[nt] = load_b_frag(sBt3[(nt << 4) + nb], lane);
    }
#pragma unroll
    for (int nt = 0; nt < 4; ++nt) {
      accG[nt] = __builtin_amdgcn_wmma_f32_16x16x32_bf16(
          false, a, false, bf1[nt], (short)0, accG[nt], false, false);
      accU[nt] = __builtin_amdgcn_wmma_f32_16x16x32_bf16(
          false, a, false, bf3[nt], (short)0, accU[nt], false, false);
    }
    __syncthreads();
  }

  // Epilogue: SwiGLU, write bf16 activations to workspace
  const int colL = lane & 15;
  const int rowH = (lane >> 4) << 3;
#pragma unroll
  for (int nt = 0; nt < 4; ++nt) {
#pragma unroll
    for (int r = 0; r < 8; ++r) {
      int mr   = (wave << 4) + rowH + r;
      int grow = row0 + mr;
      if (grow < n_e) {
        float g = accG[nt][r], u = accU[nt][r];
        float val = (g / (1.f + __expf(-g))) * u;    // silu(g) * u
        __bf16 bv = (__bf16)val;
        act[(size_t)(off + grow) * I_DIM + cb + (nt << 4) + colL] =
            __builtin_bit_cast(unsigned short, bv);
      }
    }
  }
}

// ---------------- 4) Grouped down GEMM + weighted atomic combine -----------
__global__ __launch_bounds__(256) void moe_down_kernel(
    const unsigned short* __restrict__ act, const float* __restrict__ w2,
    const int* __restrict__ offsets, const int* __restrict__ counts,
    const int* __restrict__ bucket_tok, const float* __restrict__ bucket_w,
    float* __restrict__ out) {
  const int e    = blockIdx.z;
  const int n_e  = counts[e];
  const int row0 = blockIdx.x * BM;
  if (row0 >= n_e) return;
  const int cb  = blockIdx.y * BN;       // H column base
  const int off = offsets[e];

  __shared__ unsigned int sA [BM][LDA];
  __shared__ unsigned int sBt[BN][LDA];  // transposed: [n][kpair]

  const int tid  = threadIdx.x;
  const int lane = tid & 31;
  const int wave = tid >> 5;

  // ---- Hoisted source pointers ----------------------------------------
  const int aRow0 = tid >> 2;              // + 64*it
  const int q     = (tid & 3) << 2;        // uint index 0,4,8,12
  const unsigned short* pAct[2];
#pragma unroll
  for (int it = 0; it < 2; ++it) {
    int grow = row0 + aRow0 + it * 64;
    pAct[it] = (grow < n_e)
                   ? act + (size_t)(off + grow) * I_DIM + (q << 1)
                   : nullptr;
  }
  const int k2 = tid >> 4;
  const int n4 = (tid & 15) << 2;
  const float* b2p = w2 + ((size_t)e * I_DIM + 2 * k2) * H_DIM + cb + n4;

  const v8f vzero = {0.f, 0.f, 0.f, 0.f, 0.f, 0.f, 0.f, 0.f};
  v8f acc[4];
#pragma unroll
  for (int nt = 0; nt < 4; ++nt) acc[nt] = vzero;

  for (int k0 = 0; k0 < I_DIM; k0 += BK) {
    // A tile: activations already bf16 -> copy as uint4 (8 bf16 per load)
#pragma unroll
    for (int it = 0; it < 2; ++it) {
      uint4 v = make_uint4(0u, 0u, 0u, 0u);
      if (pAct[it]) v = *(const uint4*)(pAct[it] + k0);
      *(uint4*)&sA[aRow0 + it * 64][q] = v;
    }
    // B tile: w2 32x64 fp32 -> transposed packed bf16 pairs
    {
      const float* p2 = b2p + (size_t)k0 * H_DIM;
      float4 r0 = *(const float4*)(p2);
      float4 r1 = *(const float4*)(p2 + H_DIM);
      sBt[n4 + 0][k2] = pack_bf16x2(r0.x, r1.x);
      sBt[n4 + 1][k2] = pack_bf16x2(r0.y, r1.y);
      sBt[n4 + 2][k2] = pack_bf16x2(r0.z, r1.z);
      sBt[n4 + 3][k2] = pack_bf16x2(r0.w, r1.w);
      if (k0 + BK < I_DIM)
        __builtin_prefetch(p2 + (size_t)BK * H_DIM, 0, 0);
    }
    __syncthreads();

    v16bf a = load_a_frag(sA[(wave << 4) + (lane & 15)], lane);
    const int nb = lane & 15;
    v16bf bf[4];
#pragma unroll
    for (int nt = 0; nt < 4; ++nt)
      bf[nt] = load_b_frag(sBt[(nt << 4) + nb], lane);
#pragma unroll
    for (int nt = 0; nt < 4; ++nt)
      acc[nt] = __builtin_amdgcn_wmma_f32_16x16x32_bf16(
          false, a, false, bf[nt], (short)0, acc[nt], false, false);
    __syncthreads();
  }

  // Epilogue: y[tok] += routing_weight * tile  (f32 atomics)
  const int colL = lane & 15;
  const int rowH = (lane >> 4) << 3;
#pragma unroll
  for (int nt = 0; nt < 4; ++nt) {
#pragma unroll
    for (int r = 0; r < 8; ++r) {
      int mr   = (wave << 4) + rowH + r;
      int grow = row0 + mr;
      if (grow < n_e) {
        int   slot = off + grow;
        int   tok  = bucket_tok[slot];
        float wgt  = bucket_w[slot];
        atomicAdd(out + (size_t)tok * H_DIM + cb + (nt << 4) + colL,
                  wgt * acc[nt][r]);
      }
    }
  }
}

// ---------------- Launch ----------------------------------------------------
extern "C" void kernel_launch(void* const* d_in, const int* in_sizes, int n_in,
                              void* d_out, int out_size, void* d_ws, size_t ws_size,
                              hipStream_t stream) {
  const float* x  = (const float*)d_in[0];   // [1,1024,2048]
  const float* gw = (const float*)d_in[1];   // [64,2048]
  const float* w1 = (const float*)d_in[2];   // [64,2048,768]
  const float* w3 = (const float*)d_in[3];   // [64,2048,768]
  const float* w2 = (const float*)d_in[4];   // [64,768,2048]
  float* out = (float*)d_out;

  const int T  = in_sizes[0] / H_DIM;        // 1024 tokens
  const int TK = T * K_TOP;                  // 8192 (token,expert) slots

  // Workspace carve (≈12.7 MB total)
  char* ws = (char*)d_ws;
  size_t o = 0;
  auto carve = [&](size_t bytes) -> char* {
    char* p = ws + o;
    o += (bytes + 255) & ~(size_t)255;
    return p;
  };
  int*   topk_idx   = (int*)  carve((size_t)TK * 4);
  float* topk_w     = (float*)carve((size_t)TK * 4);
  int*   counts     = (int*)  carve(E_NUM * 4);
  int*   offsets    = (int*)  carve((E_NUM + 1) * 4);
  int*   cursors    = (int*)  carve(E_NUM * 4);
  int*   bucket_tok = (int*)  carve((size_t)TK * 4);
  float* bucket_w   = (float*)carve((size_t)TK * 4);
  unsigned short* act = (unsigned short*)carve((size_t)TK * I_DIM * 2);

  hipMemsetAsync(out, 0, (size_t)out_size * sizeof(float), stream);
  hipMemsetAsync(counts, 0, E_NUM * sizeof(int), stream);

  gate_topk_kernel<<<T, E_NUM, 0, stream>>>(x, gw, topk_idx, topk_w);
  count_kernel<<<(TK + 255) / 256, 256, 0, stream>>>(topk_idx, counts, TK);
  offsets_kernel<<<1, 64, 0, stream>>>(counts, offsets, cursors);
  scatter_kernel<<<(TK + 255) / 256, 256, 0, stream>>>(
      topk_idx, topk_w, cursors, bucket_tok, bucket_w, TK);

  const int rowTiles = (T + BM - 1) / BM;    // covers worst-case expert load
  dim3 gridUp(rowTiles, I_DIM / BN, E_NUM);
  moe_up_kernel<<<gridUp, 256, 0, stream>>>(
      x, w1, w3, offsets, counts, bucket_tok, act);
  dim3 gridDown(rowTiles, H_DIM / BN, E_NUM);
  moe_down_kernel<<<gridDown, 256, 0, stream>>>(
      act, w2, offsets, counts, bucket_tok, bucket_w, out);
}
